// GuidedLocalEnhancement_67551245631507
// MI455X (gfx1250) — matrix operand order repaired and checked by reading, not compiled, not run on hardware
//
#include <hip/hip_runtime.h>
#include <hip/hip_bf16.h>

typedef __attribute__((ext_vector_type(16))) _Float16 v16h;
typedef __attribute__((ext_vector_type(8)))  float    v8f;

#define CDIM    192
#define WPB     16          // windows per block in attn kernel
#define TOKROWS (WPB * 5)   // 80
#define NQKV    576
#define A_STRIDE 208        // f16 elements, padded (416 B row = 26 * 16B)
#define Q_STRIDE 584        // f32 elements, padded
#define LN_EPS  1e-5f

// ---------------------------------------------------------------- kernel 0
__global__ void convert_w_kernel(const float* __restrict__ qkv_w,
                                 _Float16* __restrict__ wf16) {
    int idx = blockIdx.x * 256 + threadIdx.x;
    if (idx < NQKV * CDIM) wf16[idx] = (_Float16)qkv_w[idx];
}

// ---------------------------------------------------------------- kernel 1
// BN -> GELU -> depthwise 7x7 stride-2 conv. One block per (b, c) plane.
// Writes guide token-major: guide[(b*784 + oh*28 + ow)*192 + c]
__global__ __launch_bounds__(256) void guide_kernel(
    const float* __restrict__ x,
    const float* __restrict__ bn_g, const float* __restrict__ bn_b,
    const float* __restrict__ bn_m, const float* __restrict__ bn_v,
    const float* __restrict__ conv_w, const float* __restrict__ conv_b,
    float* __restrict__ guide) {
    __shared__ float tile[61 * 64];   // input span: ih,iw in [-3,57] -> 61x61
    __shared__ float wts[49];

    const int bc = blockIdx.x;
    const int c  = bc % CDIM;
    const int b  = bc / CDIM;
    const float* xp = x + (size_t)bc * 3136;

    if (threadIdx.x < 49) wts[threadIdx.x] = conv_w[c * 49 + threadIdx.x];

    const float mean = bn_m[c];
    const float inv  = rsqrtf(bn_v[c] + LN_EPS);
    const float ga   = bn_g[c];
    const float be   = bn_b[c];

    for (int idx = threadIdx.x; idx < 61 * 61; idx += 256) {
        const int ty = idx / 61, tx = idx % 61;
        const int iy = ty - 3,   ix = tx - 3;
        float v = 0.0f;
        if (iy >= 0 && iy < 56 && ix >= 0 && ix < 56) {
            float t = xp[iy * 56 + ix];
            t = (t - mean) * inv * ga + be;
            v = 0.5f * t * (1.0f + erff(t * 0.70710678118654752f)); // exact GELU
        }
        tile[ty * 64 + tx] = v;
    }
    __syncthreads();

    const float cb = conv_b[c];
    for (int idx = threadIdx.x; idx < 784; idx += 256) {
        const int oh = idx / 28, ow = idx % 28;
        float acc = cb;
        const float* tr = &tile[(2 * oh) * 64 + 2 * ow];
#pragma unroll
        for (int ky = 0; ky < 7; ++ky)
#pragma unroll
            for (int kx = 0; kx < 7; ++kx)
                acc += tr[ky * 64 + kx] * wts[ky * 7 + kx];
        guide[((size_t)b * 784 + idx) * CDIM + c] = acc;
    }
}

// ---------------------------------------------------------------- kernel 2
// Fused: token gather + LN0 -> WMMA QKV GEMM -> 1x5 attention -> LN1 -> out.
__global__ __launch_bounds__(256, 1) void attn_kernel(
    const float*    __restrict__ x,
    const float*    __restrict__ guide,
    const _Float16* __restrict__ wf16,     // [576][192] row-major (= B^T)
    const float*    __restrict__ qkv_b,
    const float*    __restrict__ ln0_g, const float* __restrict__ ln0_b,
    const float*    __restrict__ ln1_g, const float* __restrict__ ln1_b,
    float*          __restrict__ out) {
    extern __shared__ char smem[];
    _Float16* aTok = (_Float16*)smem;                                        // 80 x 208 f16
    float*    qkvS = (float*)(smem + TOKROWS * A_STRIDE * 2);                // 80 x 584 f32
    float*    oBuf = (float*)(smem + TOKROWS * A_STRIDE * 2
                                   + TOKROWS * Q_STRIDE * 4);                // 16 x 192 f32

    const int tid   = threadIdx.x;
    const int wave  = tid >> 5;
    const int lane  = tid & 31;
    const int wbase = blockIdx.x * WPB;

    // ---------------- Phase 1: gather tokens, LN0, store f16 A-matrix ----
    for (int slot = wave; slot < TOKROWS; slot += 8) {
        const int wi = slot / 5, t = slot % 5;
        const int w  = wbase + wi;
        const int b  = w / 784, p = w % 784;
        const int oh = p / 28,  ow = p % 28;

        float vals[6];
        float s = 0.0f, s2 = 0.0f;
        if (t == 0) {
            const float* gp = guide + (size_t)w * CDIM;
#pragma unroll
            for (int j = 0; j < 6; ++j) {
                float v = gp[lane + 32 * j];
                vals[j] = v; s += v; s2 += v * v;
            }
        } else {
            const int dy = (t - 1) >> 1, dx = (t - 1) & 1;
            const float* xp = x + (size_t)b * CDIM * 3136
                                + (2 * oh + dy) * 56 + (2 * ow + dx);
#pragma unroll
            for (int j = 0; j < 6; ++j) {
                float v = xp[(size_t)(lane + 32 * j) * 3136];
                vals[j] = v; s += v; s2 += v * v;
            }
        }
#pragma unroll
        for (int m = 16; m >= 1; m >>= 1) {
            s  += __shfl_xor(s,  m, 32);
            s2 += __shfl_xor(s2, m, 32);
        }
        const float mean = s * (1.0f / 192.0f);
        const float var  = s2 * (1.0f / 192.0f) - mean * mean;
        const float inv  = rsqrtf(var + LN_EPS);
#pragma unroll
        for (int j = 0; j < 6; ++j) {
            const int cidx = lane + 32 * j;
            const float y = (vals[j] - mean) * inv * ln0_g[cidx] + ln0_b[cidx];
            aTok[slot * A_STRIDE + cidx] = (_Float16)y;
        }
    }
    __syncthreads();

    // ---------------- Phase 2: QKV GEMM, v_wmma_f32_16x16x32_f16 ---------
    // Per-lane fragment bytes are two contiguous 16B runs, so frags load as
    // 2x b128 each (ds_load_b128 for A, global_load_b128 for B).
    // A 16-bit 16x32: lane half hv holds K = k0+8hv..+7 and k0+16+8hv..+7.
    // B 16-bit 32x16: lane half hv holds K = k0+8hv..+7 and k0+16+8hv..+7.
    // Two N-tiles share one A fragment -> two independent wmma chains.
    {
        const int hv  = lane >> 4;   // lane half
        const int l16 = lane & 15;
        for (int pairId = wave; pairId < 5 * (NQKV / 32); pairId += 8) {
            const int mt = pairId % 5, ntp = pairId / 5;
            const int row0 = mt * 16, n0 = ntp * 32;
            const _Float16* arow  = aTok + (size_t)(row0 + l16) * A_STRIDE;
            const _Float16* brow0 = wf16 + (size_t)(n0 + l16) * CDIM;
            const _Float16* brow1 = wf16 + (size_t)(n0 + 16 + l16) * CDIM;
            __builtin_prefetch(brow0, 0, 1);   // global_prefetch_b8
            __builtin_prefetch(brow1, 0, 1);

            union { v16h v; uint4 q[2]; } fa, fb0, fb1;
            v8f acc0 = {}, acc1 = {};
#pragma unroll
            for (int ks = 0; ks < 6; ++ks) {
                const int k0 = ks * 32;
                fa.q[0]  = *reinterpret_cast<const uint4*>(arow  + k0 + 8 * hv);
                fa.q[1]  = *reinterpret_cast<const uint4*>(arow  + k0 + 16 + 8 * hv);
                fb0.q[0] = *reinterpret_cast<const uint4*>(brow0 + k0 + 8 * hv);
                fb0.q[1] = *reinterpret_cast<const uint4*>(brow0 + k0 + 16 + 8 * hv);
                fb1.q[0] = *reinterpret_cast<const uint4*>(brow1 + k0 + 8 * hv);
                fb1.q[1] = *reinterpret_cast<const uint4*>(brow1 + k0 + 16 + 8 * hv);
                acc0 = __builtin_amdgcn_wmma_f32_16x16x32_f16(
                    false, fa.v, false, fb0.v, (short)0, acc0, false, false);
                acc1 = __builtin_amdgcn_wmma_f32_16x16x32_f16(
                    false, fa.v, false, fb1.v, (short)0, acc1, false, false);
            }
            // D layout: VGPR i -> M = row0 + i + 8*hv, N = n0(+16) + l16
            const float bias0 = qkv_b[n0 + l16];
            const float bias1 = qkv_b[n0 + 16 + l16];
#pragma unroll
            for (int i = 0; i < 8; ++i) {
                const int m = row0 + i + (hv << 3);
                qkvS[m * Q_STRIDE + n0 + l16]      = acc0[i] + bias0;
                qkvS[m * Q_STRIDE + n0 + 16 + l16] = acc1[i] + bias1;
            }
        }
    }
    __syncthreads();

    // ---------------- Phase 3: attention, only the guide query survives --
    if (tid < WPB * 8) {
        const int wi = tid >> 3, h = tid & 7;
        const float* qrow = qkvS + (wi * 5) * Q_STRIDE + h * 24;
        float sc[5];
#pragma unroll
        for (int j = 0; j < 5; ++j) {
            const float* krow = qkvS + (wi * 5 + j) * Q_STRIDE + 192 + h * 24;
            float d = 0.0f;
#pragma unroll
            for (int e = 0; e < 24; ++e) d += qrow[e] * krow[e];
            sc[j] = d * 0.20412414523193154f;   // 24^-0.5
        }
        float mx = sc[0];
#pragma unroll
        for (int j = 1; j < 5; ++j) mx = fmaxf(mx, sc[j]);
        float se = 0.0f;
#pragma unroll
        for (int j = 0; j < 5; ++j) { sc[j] = __expf(sc[j] - mx); se += sc[j]; }
        const float rs = 1.0f / se;
#pragma unroll
        for (int e = 0; e < 24; ++e) {
            float o = 0.0f;
#pragma unroll
            for (int j = 0; j < 5; ++j)
                o += sc[j] * qkvS[(wi * 5 + j) * Q_STRIDE + 384 + h * 24 + e];
            oBuf[wi * CDIM + h * 24 + e] = o * rs;
        }
    }
    __syncthreads();

    // ---------------- Phase 4: LN1 + coalesced store ---------------------
    for (int wi = wave; wi < WPB; wi += 8) {
        float vals[6];
        float s = 0.0f, s2 = 0.0f;
#pragma unroll
        for (int j = 0; j < 6; ++j) {
            float v = oBuf[wi * CDIM + lane + 32 * j];
            vals[j] = v; s += v; s2 += v * v;
        }
#pragma unroll
        for (int m = 16; m >= 1; m >>= 1) {
            s  += __shfl_xor(s,  m, 32);
            s2 += __shfl_xor(s2, m, 32);
        }
        const float mean = s * (1.0f / 192.0f);
        const float var  = s2 * (1.0f / 192.0f) - mean * mean;
        const float inv  = rsqrtf(var + LN_EPS);
#pragma unroll
        for (int j = 0; j < 6; ++j) {
            const int cidx = lane + 32 * j;
            out[((size_t)(wbase + wi)) * CDIM + cidx] =
                (vals[j] - mean) * inv * ln1_g[cidx] + ln1_b[cidx];
        }
    }
}

// ---------------------------------------------------------------- launch
extern "C" void kernel_launch(void* const* d_in, const int* in_sizes, int n_in,
                              void* d_out, int out_size, void* d_ws, size_t ws_size,
                              hipStream_t stream) {
    (void)in_sizes; (void)n_in; (void)out_size; (void)ws_size;
    const float* x        = (const float*)d_in[0];
    const float* bn_gamma = (const float*)d_in[1];
    const float* bn_beta  = (const float*)d_in[2];
    const float* bn_mean  = (const float*)d_in[3];
    const float* bn_var   = (const float*)d_in[4];
    const float* conv_w   = (const float*)d_in[5];
    const float* conv_b   = (const float*)d_in[6];
    const float* qkv_w    = (const float*)d_in[7];
    const float* qkv_b    = (const float*)d_in[8];
    const float* ln0_g    = (const float*)d_in[9];
    const float* ln0_b    = (const float*)d_in[10];
    const float* ln1_g    = (const float*)d_in[11];
    const float* ln1_b    = (const float*)d_in[12];
    float* out = (float*)d_out;

    char* ws = (char*)d_ws;
    float*    guide = (float*)ws;                            // 25088*192 f32 = 19,267,584 B
    _Float16* wf16  = (_Float16*)(ws + 25088ull * CDIM * 4); // 576*192 f16

    convert_w_kernel<<<(NQKV * CDIM + 255) / 256, 256, 0, stream>>>(qkv_w, wf16);

    guide_kernel<<<32 * CDIM, 256, 0, stream>>>(
        x, bn_gamma, bn_beta, bn_mean, bn_var, conv_w, conv_b, guide);

    const size_t smem = (size_t)TOKROWS * A_STRIDE * 2   // A tokens f16
                      + (size_t)TOKROWS * Q_STRIDE * 4   // QKV f32
                      + (size_t)WPB * CDIM * 4;          // attention out f32
    attn_kernel<<<25088 / WPB, 256, smem, stream>>>(
        x, guide, wf16, qkv_b, ln0_g, ln0_b, ln1_g, ln1_b, out);
}
